// GPT_20349555048886
// MI455X (gfx1250) — matrix operand order repaired and checked by reading, not compiled, not run on hardware
//
#include <hip/hip_runtime.h>
#include <cstddef>
#include <cstdint>

// ---------------- constants ----------------
#define TT 2048
#define DD 768
#define HN 12
#define DH 64
#define NL 12
#define FF 3072
#define VV 50257

typedef __bf16 v16bf __attribute__((ext_vector_type(16)));
typedef float  v8f   __attribute__((ext_vector_type(8)));
typedef unsigned int v4u __attribute__((ext_vector_type(4)));
typedef int v8i __attribute__((ext_vector_type(8)));
typedef int v4i __attribute__((ext_vector_type(4)));

#if defined(__has_builtin)
#if __has_builtin(__builtin_amdgcn_tensor_load_to_lds)
#define HAVE_TDM 1
#endif
#endif
#ifndef HAVE_TDM
#define HAVE_TDM 0
#endif

__device__ inline v8f wmma_bf16(v16bf a, v16bf b, v8f c) {
  return __builtin_amdgcn_wmma_f32_16x16x32_bf16(false, a, false, b, (short)0, c, false, false);
}

// ---- fragment loaders (contiguous k order k = 16*(lane/16) + s, same for A and B) ----
// A 16x32 bf16: lane m = lane&15; lanes 0-15 hold K 0-15, lanes 16-31 hold K 16-31,
// packed 2 per VGPR.
__device__ inline v16bf load_afrag(const __bf16* base, int stride, int lane) {
  union { v16bf v; unsigned u[8]; } f;
  int m = lane & 15, kh = lane >> 4;
  const __bf16* p = base + (size_t)m * stride;
#pragma unroll
  for (int i = 0; i < 8; ++i) {
    int k = kh * 16 + (i << 1);
    f.u[i] = *reinterpret_cast<const unsigned*>(p + k);
  }
  return f.v;
}

// B 32x16 bf16, element (k,n) at base + k*stride + n (row-major K x N).
// lanes 0-15: K=0..15 (n=lane), lanes 16-31: K=16..31.
__device__ inline v16bf load_bfrag_rows(const __bf16* base, int stride, int lane) {
  union { v16bf v; __bf16 e[16]; } f;
  int n = lane & 15;
  int kh = (lane >> 4) << 4;
#pragma unroll
  for (int i = 0; i < 16; ++i) f.e[i] = base[(size_t)(kh + i) * stride + n];
  return f.v;
}

// B 32x16 bf16 where consecutive K are contiguous: element (k,n) at base + n*stride + k.
__device__ inline v16bf load_bfrag_packed(const __bf16* base, int stride, int lane) {
  union { v16bf v; unsigned u[8]; } f;
  int n = lane & 15;
  int kh = (lane >> 4) << 4;
  const __bf16* p = base + (size_t)n * stride + kh;
#pragma unroll
  for (int i = 0; i < 8; ++i) f.u[i] = *reinterpret_cast<const unsigned*>(p + 2 * i);
  return f.v;
}

#if HAVE_TDM
// ---- Tensor Data Mover: 2D tile Global -> LDS (D# per CDNA5 ISA ch.8) ----
// dims/strides in elements (data_size = 2 bytes). pad inserts (pad_amt+1) dwords
// every 2^(pad_int+1) dwords into the LDS destination.
__device__ inline void tdm_load_2d(unsigned lds_addr, const void* gaddr,
                                   unsigned tensor_d0, unsigned tensor_d1,
                                   unsigned stride_d0,
                                   unsigned tile_d0, unsigned tile_d1,
                                   unsigned pad_int, unsigned pad_amt) {
  unsigned long long ga = (unsigned long long)(uintptr_t)gaddr;
  v4u g0;
  g0.x = 1u;                                       // count=1, user descriptor
  g0.y = lds_addr;                                 // lds_addr [63:32]
  g0.z = (unsigned)(ga & 0xffffffffu);             // global_addr [95:64]
  g0.w = (unsigned)((ga >> 32) & 0x01ffffffu)      // global_addr [120:96]
       | 0x80000000u;                              // type=2 ("image") [127:126]
  v8i g1;
  g1[0] = (int)((1u << 16)                         // data_size = 2 bytes
              | (1u << 20)                         // pad_enable
              | (pad_int << 22) | (pad_amt << 25));
  g1[1] = (int)((tensor_d0 & 0xffffu) << 16);      // tensor_dim0[15:0] (barrier addr = 0)
  g1[2] = (int)(((tensor_d0 >> 16) & 0xffffu) | ((tensor_d1 & 0xffffu) << 16));
  g1[3] = (int)(((tensor_d1 >> 16) & 0xffffu) | ((tile_d0 & 0xffffu) << 16));
  g1[4] = (int)(tile_d1 & 0xffffu);                // tile_dim1; tile_dim2 = 0
  g1[5] = (int)stride_d0;                          // tensor_dim0_stride[31:0]
  g1[6] = 0;                                       // stride hi / dim1_stride lo
  g1[7] = 0;
  v4i gz4 = {0, 0, 0, 0};
  v8i gz8 = {0, 0, 0, 0, 0, 0, 0, 0};
  __builtin_amdgcn_tensor_load_to_lds(g0, g1, gz4, gz4, gz8, 0);
}
#endif

// ---------------- fp32 -> bf16 conversion ----------------
__global__ __launch_bounds__(256) void f2bf_k(const float* __restrict__ s,
                                              __bf16* __restrict__ d, size_t n) {
  size_t i = ((size_t)blockIdx.x * 256 + threadIdx.x) * 4;
#pragma unroll
  for (int j = 0; j < 4; ++j)
    if (i + j < n) d[i + j] = (__bf16)s[i + j];
}

// ---------------- embedding ----------------
__global__ __launch_bounds__(256) void embed_k(const int* __restrict__ x,
                                               const float* __restrict__ wte,
                                               const float* __restrict__ wpe,
                                               float* __restrict__ h) {
  int t = blockIdx.x, tid = threadIdx.x;
  int tok = x[t];
#pragma unroll
  for (int i = 0; i < 3; ++i) {
    int d = tid + i * 256;
    h[(size_t)t * DD + d] = wte[(size_t)tok * DD + d] + wpe[(size_t)t * DD + d];
  }
}

// ---------------- layernorm fp32 -> bf16 ----------------
__global__ __launch_bounds__(256) void layernorm_k(const float* __restrict__ X,
                                                   const float* __restrict__ g,
                                                   const float* __restrict__ b,
                                                   __bf16* __restrict__ Y) {
  __shared__ float red[256];
  int row = blockIdx.x, tid = threadIdx.x;
  const float* x = X + (size_t)row * DD;
  float v0 = x[tid], v1 = x[tid + 256], v2 = x[tid + 512];
  red[tid] = v0 + v1 + v2;
  __syncthreads();
  for (int o = 128; o > 0; o >>= 1) { if (tid < o) red[tid] += red[tid + o]; __syncthreads(); }
  float mu = red[0] * (1.0f / DD);
  __syncthreads();
  float d0 = v0 - mu, d1 = v1 - mu, d2 = v2 - mu;
  red[tid] = d0 * d0 + d1 * d1 + d2 * d2;
  __syncthreads();
  for (int o = 128; o > 0; o >>= 1) { if (tid < o) red[tid] += red[tid + o]; __syncthreads(); }
  float rstd = rsqrtf(red[0] * (1.0f / DD) + 1e-5f);
  __bf16* y = Y + (size_t)row * DD;
  y[tid]       = (__bf16)(d0 * rstd * g[tid]       + b[tid]);
  y[tid + 256] = (__bf16)(d1 * rstd * g[tid + 256] + b[tid + 256]);
  y[tid + 512] = (__bf16)(d2 * rstd * g[tid + 512] + b[tid + 512]);
}

// ---------------- WMMA GEMM: C[MxN] = A[MxK](bf16) * B(bf16) (+bias,+gelu,+resid) ----
// BT=0: B is K x N row-major (LDS tile [k][n], stride 66).
// BT=1: B is N x K row-major (lm head / wte^T; LDS tile [n][k], stride 34).
#define BM 128
#define BN 64
#define BK 32
#define LDA_S 34
#define LDB_S 66
#define LDBT_S 34

template <int OUT_BF16, int GELU, int RESID, int BT>
__global__ __launch_bounds__(256) void gemm_k(const __bf16* __restrict__ A,
                                              const __bf16* __restrict__ B,
                                              const float* __restrict__ bias,
                                              const float* __restrict__ resid,
                                              float* __restrict__ Cf,
                                              __bf16* __restrict__ Cb,
                                              int M, int N, int K) {
  constexpr int NBUF = HAVE_TDM ? 2 : 1;
  constexpr int BELEM = BT ? (BN * LDBT_S) : (BK * LDB_S);
  __shared__ __bf16 sA[NBUF][BM * LDA_S];
  __shared__ __bf16 sB[NBUF][BELEM];
  int tid = threadIdx.x;
  int lane = tid & 31, wid = tid >> 5;
  int m0 = blockIdx.y * BM;
  int n0 = blockIdx.x * BN;
  int wm = (wid & 3) * 32;   // wave row quadrant (4 along M)
  int wn = (wid >> 2) * 32;  // wave col half (2 along N)
  v8f acc[2][2] = {};
  const int nsteps = K / BK;

#if HAVE_TDM
  auto issue = [&](int buf, int k0) {
    // A tile: BM x BK, rows stride K; LDS stride 34 via pad (16 dwords -> +1 dword)
    tdm_load_2d((unsigned)(uintptr_t)&sA[buf][0], A + (size_t)m0 * K + k0,
                (unsigned)(K - k0), (unsigned)(M - m0), (unsigned)K,
                BK, BM, 3u, 0u);
    if (!BT) {
      // B tile: BK x BN, rows stride N; LDS stride 66 (32 dwords -> +1 dword)
      tdm_load_2d((unsigned)(uintptr_t)&sB[buf][0], B + (size_t)k0 * N + n0,
                  (unsigned)(N - n0), (unsigned)(K - k0), (unsigned)N,
                  BN, BK, 4u, 0u);
    } else {
      // B^T tile: BN rows x BK cols from [N,K]; tensor_dim1 = N-n0 zero-fills vocab tail
      tdm_load_2d((unsigned)(uintptr_t)&sB[buf][0], B + (size_t)n0 * K + k0,
                  (unsigned)(K - k0), (unsigned)(N - n0), (unsigned)K,
                  BK, BN, 3u, 0u);
    }
  };
  if (wid == 0) issue(0, 0);
#endif

  for (int step = 0; step < nsteps; ++step) {
    int k0 = step * BK;
#if HAVE_TDM
    int buf = step & 1;
    if (wid == 0) {
      if (step + 1 < nsteps) {
        issue(buf ^ 1, k0 + BK);
        __builtin_amdgcn_s_wait_tensorcnt(2);
      } else {
        __builtin_amdgcn_s_wait_tensorcnt(0);
      }
    }
    __syncthreads();
#else
    const int buf = 0;
    __syncthreads();
    // stage A tile (128x32) as 2048 dword pairs
#pragma unroll
    for (int i = 0; i < 8; ++i) {
      int p = i * 256 + tid;
      int r = p >> 4;
      int c = (p & 15) << 1;
      unsigned val = *reinterpret_cast<const unsigned*>(A + (size_t)(m0 + r) * K + k0 + c);
      *reinterpret_cast<unsigned*>(&sA[0][r * LDA_S + c]) = val;
    }
    if (!BT) {
#pragma unroll
      for (int i = 0; i < 4; ++i) {
        int p = i * 256 + tid;
        int kr = p >> 5;
        int c = (p & 31) << 1;
        unsigned val = *reinterpret_cast<const unsigned*>(B + (size_t)(k0 + kr) * N + n0 + c);
        *reinterpret_cast<unsigned*>(&sB[0][kr * LDB_S + c]) = val;
      }
    } else {
#pragma unroll
      for (int i = 0; i < 4; ++i) {
        int p = i * 256 + tid;
        int nr = p >> 4;
        int c = (p & 15) << 1;
        unsigned val = 0u;
        if (n0 + nr < N)
          val = *reinterpret_cast<const unsigned*>(B + (size_t)(n0 + nr) * K + k0 + c);
        *reinterpret_cast<unsigned*>(&sB[0][nr * LDBT_S + c]) = val;
      }
    }
    __syncthreads();
#endif

    v16bf a0 = load_afrag(&sA[buf][(wm + 0) * LDA_S], LDA_S, lane);
    v16bf a1 = load_afrag(&sA[buf][(wm + 16) * LDA_S], LDA_S, lane);
    v16bf b0, b1;
    if (!BT) {
      b0 = load_bfrag_rows(&sB[buf][wn + 0], LDB_S, lane);
      b1 = load_bfrag_rows(&sB[buf][wn + 16], LDB_S, lane);
    } else {
      b0 = load_bfrag_packed(&sB[buf][(wn + 0) * LDBT_S], LDBT_S, lane);
      b1 = load_bfrag_packed(&sB[buf][(wn + 16) * LDBT_S], LDBT_S, lane);
    }
    acc[0][0] = wmma_bf16(a0, b0, acc[0][0]);
    acc[0][1] = wmma_bf16(a0, b1, acc[0][1]);
    acc[1][0] = wmma_bf16(a1, b0, acc[1][0]);
    acc[1][1] = wmma_bf16(a1, b1, acc[1][1]);
#if HAVE_TDM
    __syncthreads();
#endif
  }

  // epilogue: C layout = VGPR r -> row r (lanes 0-15) / r+8 (lanes 16-31), col = lane&15
#pragma unroll
  for (int mt = 0; mt < 2; ++mt)
#pragma unroll
    for (int nt = 0; nt < 2; ++nt)
#pragma unroll
      for (int r = 0; r < 8; ++r) {
        int row = m0 + wm + mt * 16 + r + ((lane >> 4) << 3);
        int col = n0 + wn + nt * 16 + (lane & 15);
        if (col < N) {
          float v = acc[mt][nt][r];
          if (bias) v += bias[col];
          if (GELU) v = 0.5f * v * (1.0f + erff(v * 0.70710678118f));
          if (RESID) v += resid[(size_t)row * N + col];
          if (OUT_BF16) Cb[(size_t)row * N + col] = (__bf16)v;
          else          Cf[(size_t)row * N + col] = v;
        }
      }
}

// ---------------- flash attention (one wave, 16 queries x one head) ----------------
__global__ __launch_bounds__(32) void attn_k(const __bf16* __restrict__ Q,
                                             const __bf16* __restrict__ Kb,
                                             const __bf16* __restrict__ Vb,
                                             __bf16* __restrict__ O) {
  __shared__ __bf16 pbuf[16 * 32];
  int lane = threadIdx.x;
  int qg = blockIdx.x * 16;   // query row base
  int ho = blockIdx.y * DH;   // head feature offset
  int lastk = qg + 15;        // last valid (causal) key

  v16bf aq0 = load_afrag(Q + (size_t)qg * DD + ho, DD, lane);       // dh 0..31
  v16bf aq1 = load_afrag(Q + (size_t)qg * DD + ho + 32, DD, lane);  // dh 32..63

  v8f o[4] = {};
  float mi[8], li[8];
#pragma unroll
  for (int r = 0; r < 8; ++r) { mi[r] = -1e30f; li[r] = 0.f; }

  for (int kc = 0; kc <= lastk; kc += 32) {
    v8f s[2];
#pragma unroll
    for (int st = 0; st < 2; ++st) {
      int kb = kc + st * 16;
      if (kb <= lastk) {
        v16bf b0 = load_bfrag_packed(Kb + (size_t)kb * DD + ho, DD, lane);
        v16bf b1 = load_bfrag_packed(Kb + (size_t)kb * DD + ho + 32, DD, lane);
        v8f c = {};
        c = wmma_bf16(aq0, b0, c);
        c = wmma_bf16(aq1, b1, c);
        s[st] = c;
      } else {
        v8f c;
#pragma unroll
        for (int r = 0; r < 8; ++r) c[r] = -1e30f;
        s[st] = c;
      }
    }
    // scale + causal mask
#pragma unroll
    for (int st = 0; st < 2; ++st) {
      int key = kc + st * 16 + (lane & 15);
#pragma unroll
      for (int r = 0; r < 8; ++r) {
        int row = qg + r + ((lane >> 4) << 3);
        float v = s[st][r] * 0.125f;  // 1/sqrt(64)
        s[st][r] = (key <= row) ? v : -1e30f;
      }
    }
    // online softmax (row stats live per C-layout row; reduce across 16-lane halves)
#pragma unroll
    for (int r = 0; r < 8; ++r) {
      float t = fmaxf(s[0][r], s[1][r]);
#pragma unroll
      for (int m = 1; m < 16; m <<= 1) t = fmaxf(t, __shfl_xor(t, m, 32));
      float mnew = fmaxf(mi[r], t);
      float alpha = __expf(mi[r] - mnew);
      float p0 = __expf(s[0][r] - mnew);
      float p1 = __expf(s[1][r] - mnew);
      s[0][r] = p0; s[1][r] = p1;
      float ps = p0 + p1;
#pragma unroll
      for (int m = 1; m < 16; m <<= 1) ps += __shfl_xor(ps, m, 32);
      li[r] = li[r] * alpha + ps;
      mi[r] = mnew;
#pragma unroll
      for (int j = 0; j < 4; ++j) o[j][r] *= alpha;
    }
    // C-layout P -> LDS -> A-fragment
#pragma unroll
    for (int st = 0; st < 2; ++st)
#pragma unroll
      for (int r = 0; r < 8; ++r) {
        int row = r + ((lane >> 4) << 3);
        int col = st * 16 + (lane & 15);
        pbuf[row * 32 + col] = (__bf16)s[st][r];
      }
    __syncthreads();
    v16bf ap = load_afrag(pbuf, 32, lane);
    // P(16x32) x V(32x64)
#pragma unroll
    for (int j = 0; j < 4; ++j) {
      v16bf bv = load_bfrag_rows(Vb + (size_t)kc * DD + ho + j * 16, DD, lane);
      o[j] = wmma_bf16(ap, bv, o[j]);
    }
    __syncthreads();
  }
  // normalize and store
#pragma unroll
  for (int r = 0; r < 8; ++r) {
    float inv = 1.0f / li[r];
    int row = qg + r + ((lane >> 4) << 3);
#pragma unroll
    for (int j = 0; j < 4; ++j)
      O[(size_t)row * DD + ho + j * 16 + (lane & 15)] = (__bf16)(o[j][r] * inv);
  }
}

// ---------------- host orchestration ----------------
extern "C" void kernel_launch(void* const* d_in, const int* in_sizes, int n_in,
                              void* d_out, int out_size, void* d_ws, size_t ws_size,
                              hipStream_t stream) {
  (void)in_sizes; (void)n_in; (void)out_size; (void)ws_size;
  const int*   x     = (const int*)d_in[0];
  const float* wte   = (const float*)d_in[1];
  const float* wpe   = (const float*)d_in[2];
  const float* ln1_g = (const float*)d_in[3];
  const float* ln1_b = (const float*)d_in[4];
  const float* wq    = (const float*)d_in[5];
  const float* wk    = (const float*)d_in[6];
  const float* wv    = (const float*)d_in[7];
  const float* wo    = (const float*)d_in[8];
  const float* bo    = (const float*)d_in[9];
  const float* ln2_g = (const float*)d_in[10];
  const float* ln2_b = (const float*)d_in[11];
  const float* w1    = (const float*)d_in[12];
  const float* b1    = (const float*)d_in[13];
  const float* w2    = (const float*)d_in[14];
  const float* b2    = (const float*)d_in[15];
  const float* lnf_g = (const float*)d_in[16];
  const float* lnf_b = (const float*)d_in[17];
  float* out = (float*)d_out;

  char* w = (char*)d_ws;
  size_t off = 0;
  auto alloc = [&](size_t bytes) { char* p = w + off; off += (bytes + 255) & ~(size_t)255; return p; };

  __bf16* wq_b  = (__bf16*)alloc((size_t)NL * DD * DD * 2);
  __bf16* wk_b  = (__bf16*)alloc((size_t)NL * DD * DD * 2);
  __bf16* wv_b  = (__bf16*)alloc((size_t)NL * DD * DD * 2);
  __bf16* wo_b  = (__bf16*)alloc((size_t)NL * DD * DD * 2);
  __bf16* w1_b  = (__bf16*)alloc((size_t)NL * DD * FF * 2);
  __bf16* w2_b  = (__bf16*)alloc((size_t)NL * FF * DD * 2);
  __bf16* wte_b = (__bf16*)alloc((size_t)VV * DD * 2);
  float*  h     = (float*)alloc((size_t)TT * DD * 4);
  __bf16* xb    = (__bf16*)alloc((size_t)TT * DD * 2);
  __bf16* qb    = (__bf16*)alloc((size_t)TT * DD * 2);
  __bf16* kb    = (__bf16*)alloc((size_t)TT * DD * 2);
  __bf16* vb    = (__bf16*)alloc((size_t)TT * DD * 2);
  __bf16* ab    = (__bf16*)alloc((size_t)TT * DD * 2);
  __bf16* m1    = (__bf16*)alloc((size_t)TT * FF * 2);

  auto conv = [&](const float* s, __bf16* d, size_t n) {
    int blocks = (int)((n + 1023) / 1024);
    f2bf_k<<<blocks, 256, 0, stream>>>(s, d, n);
  };
  conv(wq,  wq_b,  (size_t)NL * DD * DD);
  conv(wk,  wk_b,  (size_t)NL * DD * DD);
  conv(wv,  wv_b,  (size_t)NL * DD * DD);
  conv(wo,  wo_b,  (size_t)NL * DD * DD);
  conv(w1,  w1_b,  (size_t)NL * DD * FF);
  conv(w2,  w2_b,  (size_t)NL * FF * DD);
  conv(wte, wte_b, (size_t)VV * DD);

  embed_k<<<TT, 256, 0, stream>>>(x, wte, wpe, h);

  dim3 g_d(DD / BN, TT / BM);   // 12 x 16
  dim3 g_f(FF / BN, TT / BM);   // 48 x 16
  dim3 g_at(TT / 16, HN);       // 128 x 12

  for (int l = 0; l < NL; ++l) {
    layernorm_k<<<TT, 256, 0, stream>>>(h, ln1_g + (size_t)l * DD, ln1_b + (size_t)l * DD, xb);
    gemm_k<1, 0, 0, 0><<<g_d, 256, 0, stream>>>(xb, wq_b + (size_t)l * DD * DD, nullptr, nullptr, nullptr, qb, TT, DD, DD);
    gemm_k<1, 0, 0, 0><<<g_d, 256, 0, stream>>>(xb, wk_b + (size_t)l * DD * DD, nullptr, nullptr, nullptr, kb, TT, DD, DD);
    gemm_k<1, 0, 0, 0><<<g_d, 256, 0, stream>>>(xb, wv_b + (size_t)l * DD * DD, nullptr, nullptr, nullptr, vb, TT, DD, DD);
    attn_k<<<g_at, 32, 0, stream>>>(qb, kb, vb, ab);
    gemm_k<0, 0, 1, 0><<<g_d, 256, 0, stream>>>(ab, wo_b + (size_t)l * DD * DD, bo + (size_t)l * DD, h, h, nullptr, TT, DD, DD);
    layernorm_k<<<TT, 256, 0, stream>>>(h, ln2_g + (size_t)l * DD, ln2_b + (size_t)l * DD, xb);
    gemm_k<1, 1, 0, 0><<<g_f, 256, 0, stream>>>(xb, w1_b + (size_t)l * DD * FF, b1 + (size_t)l * FF, nullptr, nullptr, m1, TT, FF, DD);
    gemm_k<0, 0, 1, 0><<<g_d, 256, 0, stream>>>(m1, w2_b + (size_t)l * FF * DD, b2 + (size_t)l * DD, h, h, nullptr, TT, DD, FF);
  }

  layernorm_k<<<TT, 256, 0, stream>>>(h, lnf_g, lnf_b, xb);
  dim3 g_lm((VV + BN - 1) / BN, TT / BM);  // 786 x 16
  gemm_k<0, 0, 0, 1><<<g_lm, 256, 0, stream>>>(xb, wte_b, nullptr, nullptr, out, nullptr, TT, VV, DD);
}
